// bipartitegraph_encoder_44813688766724
// MI455X (gfx1250) — compile-verified, matrix operand order checked by reading, not run on hardware
//
#include <hip/hip_runtime.h>

// ---------------- problem constants ----------------
#define BB    64
#define LL    128
#define DIMM  41
#define FULLN (LL * DIMM)        // 5248
#define BF    (BB * FULLN)       // 335872
#define TIMM  64
#define NKK   128
#define NIND  32
#define NHH   4

typedef __attribute__((ext_vector_type(16))) _Float16 v16h;
typedef __attribute__((ext_vector_type(8)))  _Float16 v8h;
typedef __attribute__((ext_vector_type(8)))  float    v8f;

union frag16 { v16h v; v8h h[2]; };

// K-major fragment load (CDNA5 16-bit A layout: lanes 0-15 K base 0,
// lanes 16-31 K base 8; VGPR4-7 carry K+16). Works for global or LDS pointers.
__device__ inline v16h load_frag(const _Float16* rowp, int k0, int laneH) {
  frag16 f;
  f.h[0] = *(const v8h*)(rowp + k0 + laneH * 8);
  f.h[1] = *(const v8h*)(rowp + k0 + 16 + laneH * 8);
  return f.v;
}

// ---------------- 1. stable partition scan (valid-first permutation) ----------------
__global__ __launch_bounds__(256) void scan_partition(const float* __restrict__ mask,
                                                      int* __restrict__ perm,
                                                      float* __restrict__ mk) {
  const int b = blockIdx.x;
  const int t = threadIdx.x;
  __shared__ int sc[256];
  int running = 0;
  for (int c0 = 0; c0 < FULLN; c0 += 256) {
    int i = c0 + t;
    int m = (i < FULLN) && (mask[(long)b * FULLN + i] > 0.5f);
    sc[t] = m;
    __syncthreads();
    for (int off = 1; off < 256; off <<= 1) {
      int v = (t >= off) ? sc[t - off] : 0;
      __syncthreads();
      sc[t] += v;
      __syncthreads();
    }
    int incl = sc[t];
    int total = sc[255];
    if (m) perm[(long)b * FULLN + (running + incl - 1)] = i;
    running += total;
    __syncthreads();
  }
  const int nv = running;
  for (int d = t; d < FULLN; d += 256) {
    mk[(long)b * FULLN + d] = (d < nv) ? 1.f : 0.f;
    if (d >= nv) perm[(long)b * FULLN + d] = -1;
  }
}

// ---------------- 2. build att0 (f16, 128-wide zero-padded) + target outputs ----------------
__global__ __launch_bounds__(256) void build_att(const int* __restrict__ perm,
                                                 const float* __restrict__ ctx,
                                                 const float* __restrict__ value,
                                                 const float* __restrict__ tval,
                                                 const float* __restrict__ tmask,
                                                 const float* __restrict__ tw,
                                                 const float* __restrict__ tb,
                                                 _Float16* __restrict__ att,
                                                 float* __restrict__ out_tU,
                                                 float* __restrict__ out_tM) {
  long r = (long)blockIdx.x * 256 + threadIdx.x;
  if (r >= (long)BF) return;
  int b = (int)(r / FULLN);
  int src = perm[r];
  float T = 0.f, U = 0.f, tU = 0.f, tM = 0.f;
  int c = 0;
  if (src >= 0) {
    T = ctx[b * LL + src / DIMM];
    U = value[(long)b * FULLN + src];
    tU = tval[(long)b * FULLN + src];
    tM = tmask[(long)b * FULLN + src];
    c = src % DIMM;
  }
  _Float16* arow = att + r * 128;
  #pragma unroll 4
  for (int j = 0; j < TIMM; j++) arow[j] = (_Float16)__sinf(T * tw[j] + tb[j]);
  for (int j = 0; j < DIMM; j++) arow[TIMM + j] = (_Float16)((j == c) ? 1.f : 0.f);
  arow[105] = (_Float16)U;
  arow[106] = (_Float16)tM;
  for (int j = 107; j < 128; j++) arow[j] = (_Float16)0.f;
  out_tU[r] = tU;
  out_tM[r] = tM;
}

// ---------------- 3. weight prep: f32 [fi,128] -> f16 N-major [128][128], K zero-padded ----
__global__ __launch_bounds__(256) void prep_w(const float* __restrict__ W,
                                              _Float16* __restrict__ Wt, int fi) {
  int idx = blockIdx.x * 256 + threadIdx.x;   // 0..16383
  int n = idx >> 7, k = idx & 127;
  Wt[n * 128 + k] = (k < fi) ? (_Float16)W[k * 128 + n] : (_Float16)0.f;
}

__global__ __launch_bounds__(256) void ind_init(_Float16* __restrict__ ind) {
  int idx = blockIdx.x * 256 + threadIdx.x;   // B*32*128
  int q = (idx >> 7) & 31, k = idx & 127;
  ind[idx] = (_Float16)((k == q) ? 1.f : 0.f);
}

// ---------------- 4. batched WMMA GEMM: Y = act(X @ W + bias) [* mk] ----------------
// X: [B, M, 128] f16, Wt: [128][128] f16 N-major, Y: [B, M, 128] f16.
// Wt is staged once per block into LDS via gfx1250 async load-to-LDS; B fragments
// then come from ds_load_b128 while A fragments for all 4 K-steps are prefetched
// from global before the barrier (one loadcnt wait covers the whole MAC chain).
__global__ __launch_bounds__(256) void gemm128_wmma(const _Float16* __restrict__ X,
                                                    const _Float16* __restrict__ Wt,
                                                    const float* __restrict__ bias,
                                                    _Float16* __restrict__ Y,
                                                    const float* __restrict__ mkrow,
                                                    int M, int relu) {
  __shared__ __align__(16) _Float16 Wl[128 * 128];   // 32 KB
  const int tid = threadIdx.x;
  const int b = blockIdx.y;
  const int wave = tid >> 5;
  const int lane = tid & 31;
  const int rowTile = blockIdx.x * 64 + (wave & 3) * 16;
  const int colBase = (wave >> 2) * 64;
  const int laneR = lane & 15;
  const int laneH = lane >> 4;
  const long xbase = (long)b * M * 128;

  // ---- async stage of Wt into LDS (128 B per thread, 8 x b128) ----
  {
    unsigned ldsa = (unsigned)(unsigned long long)(const void*)(&Wl[0]) + (unsigned)tid * 128u;
    unsigned long long ga = (unsigned long long)Wt + (unsigned long long)tid * 128ull;
    #pragma unroll
    for (int i = 0; i < 8; i++) {
      asm volatile("global_load_async_to_lds_b128 %0, %1, off"
                   :: "v"(ldsa + (unsigned)(i * 16)),
                      "v"(ga + (unsigned long long)(i * 16))
                   : "memory");
    }
  }

  // ---- prefetch A fragments for all 4 K-steps (overlaps with async stage) ----
  int arow = rowTile + laneR;
  if (arow > M - 1) arow = M - 1;                 // clamp loads; stores guarded
  const _Float16* xrow = X + xbase + (long)arow * 128;
  v16h a0 = load_frag(xrow, 0, laneH);
  v16h a1 = load_frag(xrow, 32, laneH);
  v16h a2 = load_frag(xrow, 64, laneH);
  v16h a3 = load_frag(xrow, 96, laneH);

  asm volatile("s_wait_asynccnt 0x0" ::: "memory");
  __syncthreads();

  const _Float16* wl0 = &Wl[(colBase +  0 + laneR) * 128];
  const _Float16* wl1 = &Wl[(colBase + 16 + laneR) * 128];
  const _Float16* wl2 = &Wl[(colBase + 32 + laneR) * 128];
  const _Float16* wl3 = &Wl[(colBase + 48 + laneR) * 128];

  v8f acc0 = {}, acc1 = {}, acc2 = {}, acc3 = {};
  v16h afr[4] = {a0, a1, a2, a3};
  #pragma unroll
  for (int ks = 0; ks < 4; ks++) {
    const int k0 = ks * 32;
    v16h bf0 = load_frag(wl0, k0, laneH);
    v16h bf1 = load_frag(wl1, k0, laneH);
    v16h bf2 = load_frag(wl2, k0, laneH);
    v16h bf3 = load_frag(wl3, k0, laneH);
    acc0 = __builtin_amdgcn_wmma_f32_16x16x32_f16(false, afr[ks], false, bf0, (short)0, acc0, false, false);
    acc1 = __builtin_amdgcn_wmma_f32_16x16x32_f16(false, afr[ks], false, bf1, (short)0, acc1, false, false);
    acc2 = __builtin_amdgcn_wmma_f32_16x16x32_f16(false, afr[ks], false, bf2, (short)0, acc2, false, false);
    acc3 = __builtin_amdgcn_wmma_f32_16x16x32_f16(false, afr[ks], false, bf3, (short)0, acc3, false, false);
  }

  const int rowOut = rowTile + laneH * 8;
  v8f accs[4] = {acc0, acc1, acc2, acc3};
  #pragma unroll
  for (int tt = 0; tt < 4; tt++) {
    int ncol = colBase + tt * 16 + laneR;
    float bv = bias ? bias[ncol] : 0.f;
    #pragma unroll
    for (int rr = 0; rr < 8; rr++) {
      int row = rowOut + rr;
      if (row < M) {
        float v = accs[tt][rr] + bv;
        if (relu) v = fmaxf(v, 0.f);
        if (mkrow) v *= mkrow[(long)b * M + row];
        Y[xbase + (long)row * 128 + ncol] = (_Float16)v;
      }
    }
  }
}

// ---------------- 5. ind-side attention (32 queries, FULL keys, masked, online softmax) ----
__global__ __launch_bounds__(256) void ind_attention(_Float16* __restrict__ indQ,  // [B,32,128] in/out (Qf -> Qf+o)
                                                     const _Float16* __restrict__ Kb,
                                                     const _Float16* __restrict__ Vb,
                                                     const float* __restrict__ mk) {
  const int h = blockIdx.x, b = blockIdx.y;
  const int t = threadIdx.x;
  __shared__ float Qs[32][32];
  __shared__ float Ks[32][32];
  __shared__ float Vs[32][32];
  __shared__ float Ss[32][32];
  __shared__ float mrow[32], lrow[32], corr[32], mkc[32];
  const int q = t >> 3, seg = (t & 7) * 4;
  for (int x = t; x < 1024; x += 256) {
    int qq = x >> 5, dd = x & 31;
    Qs[qq][dd] = (float)indQ[((long)b * 32 + qq) * 128 + h * 32 + dd];
  }
  if (t < 32) { mrow[t] = -1e30f; lrow[t] = 0.f; }
  float acc[4] = {0.f, 0.f, 0.f, 0.f};
  const float scale = 0.17677669529663687f;   // 1/sqrt(32)
  __syncthreads();
  for (int k0 = 0; k0 < FULLN; k0 += 32) {
    for (int x = t; x < 1024; x += 256) {
      int kk = x >> 5, dd = x & 31;
      long base = ((long)b * FULLN + k0 + kk) * 128 + h * 32 + dd;
      Ks[kk][dd] = (float)Kb[base];
      Vs[kk][dd] = (float)Vb[base];
    }
    if (t < 32) mkc[t] = mk[(long)b * FULLN + k0 + t];
    __syncthreads();
    for (int kk = seg; kk < seg + 4; kk++) {
      float dot = 0.f;
      #pragma unroll
      for (int dd = 0; dd < 32; dd++) dot += Qs[q][dd] * Ks[kk][dd];
      Ss[q][kk] = (mkc[kk] > 0.f) ? dot * scale : -1e9f;
    }
    __syncthreads();
    if (t < 32) {
      float mx = mrow[t];
      for (int kk = 0; kk < 32; kk++) mx = fmaxf(mx, Ss[t][kk]);
      float c = __expf(mrow[t] - mx);
      corr[t] = c;
      float sum = 0.f;
      for (int kk = 0; kk < 32; kk++) { float p = __expf(Ss[t][kk] - mx); Ss[t][kk] = p; sum += p; }
      mrow[t] = mx;
      lrow[t] = lrow[t] * c + sum;
    }
    __syncthreads();
    float c = corr[q];
    #pragma unroll
    for (int j = 0; j < 4; j++) acc[j] *= c;
    for (int kk = 0; kk < 32; kk++) {
      float p = Ss[q][kk];
      #pragma unroll
      for (int j = 0; j < 4; j++) acc[j] += p * Vs[kk][seg + j];
    }
    __syncthreads();
  }
  float linv = 1.f / lrow[q];
  #pragma unroll
  for (int j = 0; j < 4; j++) {
    long o = ((long)b * 32 + q) * 128 + h * 32 + seg + j;
    indQ[o] = (_Float16)(acc[j] * linv + Qs[q][seg + j]);
  }
}

// ---------------- 6. att-side attention (FULL queries, 32 keys in LDS) ----------------
__global__ __launch_bounds__(256) void att_attention(_Float16* __restrict__ Qb,  // [B,FULL,128] in/out (Qf -> Qf+o)
                                                     const _Float16* __restrict__ Ki,
                                                     const _Float16* __restrict__ Vi) {
  const int b = blockIdx.y;
  const int r = blockIdx.x * 256 + threadIdx.x;
  __shared__ float Ks[NHH][32][32];
  __shared__ float Vs[NHH][32][32];
  for (int x = threadIdx.x; x < 4096; x += 256) {
    int kk = x >> 7, col = x & 127;
    long base = ((long)b * 32 + kk) * 128 + col;
    Ks[col >> 5][kk][col & 31] = (float)Ki[base];
    Vs[col >> 5][kk][col & 31] = (float)Vi[base];
  }
  __syncthreads();
  if (r >= FULLN) return;
  const long rb = ((long)b * FULLN + r) * 128;
  const float scale = 0.17677669529663687f;
  for (int h = 0; h < NHH; h++) {
    float qv[32];
    #pragma unroll
    for (int d = 0; d < 32; d++) qv[d] = (float)Qb[rb + h * 32 + d];
    float s[32];
    float mx = -1e30f;
    for (int kk = 0; kk < 32; kk++) {
      float dot = 0.f;
      #pragma unroll
      for (int d = 0; d < 32; d++) dot += qv[d] * Ks[h][kk][d];
      s[kk] = dot * scale;
      mx = fmaxf(mx, s[kk]);
    }
    float sum = 0.f;
    for (int kk = 0; kk < 32; kk++) { s[kk] = __expf(s[kk] - mx); sum += s[kk]; }
    float inv = 1.f / sum;
    #pragma unroll
    for (int d = 0; d < 32; d++) {
      float o = 0.f;
      for (int kk = 0; kk < 32; kk++) o += s[kk] * Vs[h][kk][d];
      Qb[rb + h * 32 + d] = (_Float16)(qv[d] + o * inv);
    }
  }
}

// ---------------- 7. output projection ----------------
__global__ __launch_bounds__(256) void out_proj(const _Float16* __restrict__ att,
                                                const float* __restrict__ ow,
                                                const float* __restrict__ ob,
                                                float* __restrict__ out) {
  long r = (long)blockIdx.x * 256 + threadIdx.x;
  if (r >= (long)BF) return;
  const _Float16* arow = att + r * 128;
  float s = 0.f;
  #pragma unroll 8
  for (int n = 0; n < 128; n++) s += (float)arow[n] * ow[n];
  out[r] = s + ob[0];
}

// ---------------- host launch ----------------
extern "C" void kernel_launch(void* const* d_in, const int* in_sizes, int n_in,
                              void* d_out, int out_size, void* d_ws, size_t ws_size,
                              hipStream_t stream) {
  (void)in_sizes; (void)n_in; (void)out_size; (void)ws_size;
  const float* context_x    = (const float*)d_in[0];
  const float* value        = (const float*)d_in[1];
  const float* mask         = (const float*)d_in[2];
  const float* target_value = (const float*)d_in[3];
  const float* target_mask  = (const float*)d_in[4];
  const float* time_w       = (const float*)d_in[5];
  const float* time_b       = (const float*)d_in[6];
  const float* out_w        = (const float*)d_in[55];
  const float* out_b        = (const float*)d_in[56];

  // workspace layout (all 256B aligned)
  char* ws = (char*)d_ws;
  const size_t ATT_B = (size_t)BF * 128 * sizeof(_Float16);   // 85,983,232
  _Float16* att  = (_Float16*)(ws);
  _Float16* Kb   = (_Float16*)(ws + ATT_B);                   // reused as Qbuf
  _Float16* Vb   = (_Float16*)(ws + 2 * ATT_B);
  float*    mk   = (float*)  (ws + 3 * ATT_B);
  int*      perm = (int*)    (ws + 3 * ATT_B + (size_t)BF * 4);
  _Float16* ind  = (_Float16*)(ws + 3 * ATT_B + 2 * (size_t)BF * 4);
  _Float16* indK = ind  + (size_t)BB * 32 * 128;
  _Float16* indV = indK + (size_t)BB * 32 * 128;
  _Float16* indQ = indV + (size_t)BB * 32 * 128;
  _Float16* wts  = indQ + (size_t)BB * 32 * 128;              // 24 x [128][128] f16

  float* out0 = (float*)d_out;          // output [B,FULL]
  float* outU = out0 + (size_t)BF;      // target_U_
  float* outM = outU + (size_t)BF;      // target_mask_

  // 1. permutation + mask
  scan_partition<<<BB, 256, 0, stream>>>(mask, perm, mk);
  // 2. att0 + target outputs
  build_att<<<(BF + 255) / 256, 256, 0, stream>>>(perm, context_x, value, target_value,
                                                  target_mask, time_w, time_b, att, outU, outM);
  // 3. weights -> f16 N-major, K padded to 128
  static const int fiTab[24] = {
      32, 107, 107, 128, 107, 128, 128, 128,   // layer0: mab1 q,k,v,o ; mab2 q,k,v,o
      128, 128, 128, 128, 128, 128, 128, 128,  // layer1
      128, 128, 128, 128, 128, 128, 128, 128}; // layer2
  for (int l = 0; l < 3; l++)
    for (int m = 0; m < 2; m++)
      for (int j = 0; j < 4; j++) {
        int widx = l * 8 + m * 4 + j;
        const float* src = (const float*)d_in[7 + l * 16 + m * 8 + j * 2];
        prep_w<<<64, 256, 0, stream>>>(src, wts + (size_t)widx * 128 * 128, fiTab[widx]);
      }
  // 4. ind = identity (padded to 128)
  ind_init<<<(BB * 32 * 128) / 256, 256, 0, stream>>>(ind);

  auto bias = [&](int l, int m, int j) { return (const float*)d_in[7 + l * 16 + m * 8 + j * 2 + 1]; };
  auto W    = [&](int widx) { return (const _Float16*)(wts + (size_t)widx * 128 * 128); };
  const dim3 gridBig((FULLN + 63) / 64, BB);
  const dim3 gridSmall(1, BB);

  for (int l = 0; l < 3; l++) {
    // ---- MAB1: ind attends to att ----
    gemm128_wmma<<<gridBig, 256, 0, stream>>>(att, W(l * 8 + 1), bias(l, 0, 1), Kb, nullptr, FULLN, 0);
    gemm128_wmma<<<gridBig, 256, 0, stream>>>(att, W(l * 8 + 2), bias(l, 0, 2), Vb, nullptr, FULLN, 0);
    gemm128_wmma<<<gridSmall, 256, 0, stream>>>(ind, W(l * 8 + 0), bias(l, 0, 0), indQ, nullptr, 32, 0);
    ind_attention<<<dim3(NHH, BB), 256, 0, stream>>>(indQ, Kb, Vb, mk);
    gemm128_wmma<<<gridSmall, 256, 0, stream>>>(indQ, W(l * 8 + 3), bias(l, 0, 3), ind, nullptr, 32, 1);
    // ---- MAB2: att attends to ind ----
    gemm128_wmma<<<gridBig, 256, 0, stream>>>(att, W(l * 8 + 4), bias(l, 1, 0), Kb, nullptr, FULLN, 0); // Qbuf
    gemm128_wmma<<<gridSmall, 256, 0, stream>>>(ind, W(l * 8 + 5), bias(l, 1, 1), indK, nullptr, 32, 0);
    gemm128_wmma<<<gridSmall, 256, 0, stream>>>(ind, W(l * 8 + 6), bias(l, 1, 2), indV, nullptr, 32, 0);
    att_attention<<<dim3((FULLN + 255) / 256, BB), 256, 0, stream>>>(Kb, indK, indV);
    gemm128_wmma<<<gridBig, 256, 0, stream>>>(Kb, W(l * 8 + 7), bias(l, 1, 3), att, mk, FULLN, 1);
  }
  // 7. final projection
  out_proj<<<(BF + 255) / 256, 256, 0, stream>>>(att, out_w, out_b, out0);
}
